// PCARotatedQuantizer_46162308497804
// MI455X (gfx1250) — matrix operand order repaired and checked by reading, not compiled, not run on hardware
//
#include <hip/hip_runtime.h>
#include <stdint.h>

// ---------------- CDNA5 WMMA types ----------------
typedef __attribute__((ext_vector_type(16))) __bf16        v16bf;
typedef __attribute__((ext_vector_type(8)))  float         v8f;
typedef __attribute__((ext_vector_type(4)))  unsigned int  u32x4;
typedef __attribute__((ext_vector_type(4)))  float         f32x4;

#define DIMS 128
#define ROWS_PER_BLOCK 128
#define NTHREADS 256           // 8 wave32
#define XPITCH 136             // X/Z tile row pitch in halves: 272 B = 17*16 B
                               //  -> lane bank group (L*68 + c)%64 distinct per lane
                               //  -> conflict-free ds_load_b128 A-fragments

// ---- bf16 split helpers (round-to-nearest-even) ----
__device__ __forceinline__ unsigned short f2bf(float f) {
  unsigned u = __float_as_uint(f);
  u += 0x7FFFu + ((u >> 16) & 1u);
  return (unsigned short)(u >> 16);
}
__device__ __forceinline__ float bf2f(unsigned short h) {
  return __uint_as_float(((unsigned)h) << 16);
}

union Frag { u32x4 q[2]; v16bf v; };

// A-fragment: lane holds row (lane&15); K chunks {c0..c0+7} and {c0+16..c0+23}, c0=8*(lane>=16)
__device__ __forceinline__ v16bf load_a(const unsigned short* p) {
  Frag f;
  f.q[0] = *(const u32x4*)(p);
  f.q[1] = *(const u32x4*)(p + 16);
  return f.v;
}
// B-fragment: lane holds col (lane&15); K chunks {c0..c0+7} and {c0+16..c0+23}, c0=8*(lane>=16)
__device__ __forceinline__ v16bf load_b(const unsigned short* p) {
  Frag f;
  f.q[0] = *(const u32x4*)(p);
  f.q[1] = *(const u32x4*)(p + 16);
  return f.v;
}

#define WMMA_BF16(A, B, C) \
  __builtin_amdgcn_wmma_f32_16x16x32_bf16(false, (A), false, (B), (short)0, (C), false, false)

// ---------------- prep: split R / R^T to bf16 hi/lo, bounds, 1/ws ----------------
__global__ void pcaq_prep(const float* __restrict__ rot,
                          const float* __restrict__ cent,
                          const int*   __restrict__ bits,
                          const float* __restrict__ whiten,
                          unsigned short* __restrict__ g1h, unsigned short* __restrict__ g1l,
                          unsigned short* __restrict__ g2h, unsigned short* __restrict__ g2l,
                          float* __restrict__ bounds, float* __restrict__ invws)
{
  int t = blockIdx.x * NTHREADS + threadIdx.x;       // 0..16383
  if (t < DIMS * DIMS) {
    int i = t >> 7, j = t & (DIMS - 1);
    float r1 = rot[i * DIMS + j];                    // G1[n][k] = R[n][k]  (B = R^T)
    unsigned short h1 = f2bf(r1);
    g1h[t] = h1; g1l[t] = f2bf(r1 - bf2f(h1));
    float r2 = rot[j * DIMS + i];                    // G2[n][k] = R[k][n]  (B = R)
    unsigned short h2 = f2bf(r2);
    g2h[t] = h2; g2l[t] = f2bf(r2 - bf2f(h2));
  }
  if (t < DIMS * 64) {
    int d = t >> 6, k = t & 63;
    int Km1 = (1 << bits[d]) - 1;
    bounds[t] = (k < Km1) ? 0.5f * (cent[d * 64 + k] + cent[d * 64 + k + 1])
                          : __builtin_inff();
  }
  if (t < DIMS) invws[t] = 1.0f / whiten[t];
}

// ---------------- main fused kernel ----------------
__global__ __launch_bounds__(NTHREADS, 2)
void pcaq_main(const float* __restrict__ x,
               const float* __restrict__ mean,
               const float* __restrict__ whiten,
               const float* __restrict__ centroids,
               const unsigned short* __restrict__ g1h, const unsigned short* __restrict__ g1l,
               const unsigned short* __restrict__ g2h, const unsigned short* __restrict__ g2l,
               const float* __restrict__ bounds_g,
               const float* __restrict__ invws_g,
               float* __restrict__ out_x,
               int*   __restrict__ out_idx)
{
  // X tile (bf16 hi/lo, padded pitch) -- reused as Z tile after GEMM1 (wave-private rows)
  __shared__ __attribute__((aligned(16))) unsigned short Xhi[ROWS_PER_BLOCK * XPITCH];
  __shared__ __attribute__((aligned(16))) unsigned short Xlo[ROWS_PER_BLOCK * XPITCH];
  // stride-65 padding -> bank-conflict-free binary search (lane d stride = 65 words)
  __shared__ __attribute__((aligned(16))) float Blds[DIMS * 65];
  __shared__ __attribute__((aligned(16))) float Clds[DIMS * 65];

  const int  t       = threadIdx.x;
  const long rowBase = (long)blockIdx.x * ROWS_PER_BLOCK;

  // ---- stage centered x as bf16 hi/lo (coalesced b128 loads) ----
  {
    const f32x4* xg  = (const f32x4*)(x + rowBase * DIMS);
    const int    col = (4 * t) & (DIMS - 1);                // constant per thread
    const f32x4  mv  = *(const f32x4*)(mean + col);
    #pragma unroll
    for (int i = 0; i < (ROWS_PER_BLOCK * DIMS / 4) / NTHREADS; ++i) {
      int   f   = t + NTHREADS * i;                         // float4 index
      int   row = f >> 5;                                   // 32 float4 per row
      int   cb  = (f & 31) * 4;
      f32x4 v   = xg[f];
      #pragma unroll
      for (int j = 0; j < 4; ++j) {
        float c = v[j] - mv[j];
        unsigned short h = f2bf(c);
        Xhi[row * XPITCH + cb + j] = h;
        Xlo[row * XPITCH + cb + j] = f2bf(c - bf2f(h));
      }
    }
    for (int s = t; s < DIMS * 64; s += NTHREADS) {
      int d = s >> 6, k = s & 63;
      Blds[d * 65 + k] = bounds_g[s];
      Clds[d * 65 + k] = centroids[s];
    }
  }
  __syncthreads();

  const int lane = t & 31;
  const int wave = t >> 5;
  const int half = lane >> 4;
  const int m    = lane & 15;
  const int lrow = wave * 16 + m;          // A-fragment LDS row
  const int c0a  = half ? 8 : 0;           // A K-chunk base
  const int c0b  = half ? 8 : 0;           // B K-chunk base

  v8f acc[8] = {};

  // ---- GEMM1: Y = Xc @ R^T   (3-term bf16 split, f32 accumulate) ----
  #pragma unroll
  for (int kk = 0; kk < 4; ++kk) {
    const v16bf ah = load_a(&Xhi[lrow * XPITCH + 32 * kk + c0a]);
    const v16bf al = load_a(&Xlo[lrow * XPITCH + 32 * kk + c0a]);
    #pragma unroll
    for (int jn = 0; jn < 8; ++jn) {
      const int n = jn * 16 + m;
      const v16bf bh = load_b(g1h + n * DIMS + 32 * kk + c0b);
      const v16bf bl = load_b(g1l + n * DIMS + 32 * kk + c0b);
      acc[jn] = WMMA_BF16(ah, bh, acc[jn]);
      acc[jn] = WMMA_BF16(ah, bl, acc[jn]);
      acc[jn] = WMMA_BF16(al, bh, acc[jn]);
    }
  }

  // ---- quantize: y*ws -> searchsorted(bounds) -> idx; Z = C[idx]/ws back to LDS ----
  #pragma unroll
  for (int jn = 0; jn < 8; ++jn) {
    const int   n   = jn * 16 + m;
    const float wsn = whiten[n];
    const float iwn = invws_g[n];
    #pragma unroll
    for (int r = 0; r < 8; ++r) {
      const float y = acc[jn][r] * wsn;
      int pos = 0;                                   // lower_bound over 64 (pad = +inf)
      #pragma unroll
      for (int s = 32; s; s >>= 1)
        if (Blds[n * 65 + pos + s - 1] < y) pos += s;
      const float cv = Clds[n * 65 + pos];
      const int   lr = wave * 16 + r + 8 * half;
      out_idx[(rowBase + lr) * DIMS + n] = pos;
      const float zv = cv * iwn;
      const unsigned short zh = f2bf(zv);
      Xhi[lr * XPITCH + n] = zh;                     // wave-private rows: no barrier needed
      Xlo[lr * XPITCH + n] = f2bf(zv - bf2f(zh));
    }
  }

  // ---- GEMM2: Xhat = Z @ R + mean ----
  {
    const v8f zero = {};
    #pragma unroll
    for (int jn = 0; jn < 8; ++jn) acc[jn] = zero;
  }
  #pragma unroll
  for (int kk = 0; kk < 4; ++kk) {
    const v16bf ah = load_a(&Xhi[lrow * XPITCH + 32 * kk + c0a]);
    const v16bf al = load_a(&Xlo[lrow * XPITCH + 32 * kk + c0a]);
    #pragma unroll
    for (int jn = 0; jn < 8; ++jn) {
      const int n = jn * 16 + m;
      const v16bf bh = load_b(g2h + n * DIMS + 32 * kk + c0b);
      const v16bf bl = load_b(g2l + n * DIMS + 32 * kk + c0b);
      acc[jn] = WMMA_BF16(ah, bh, acc[jn]);
      acc[jn] = WMMA_BF16(ah, bl, acc[jn]);
      acc[jn] = WMMA_BF16(al, bh, acc[jn]);
    }
  }
  #pragma unroll
  for (int jn = 0; jn < 8; ++jn) {
    const int   n  = jn * 16 + m;
    const float mn = mean[n];
    #pragma unroll
    for (int r = 0; r < 8; ++r) {
      const long grow = rowBase + wave * 16 + r + 8 * half;
      out_x[grow * DIMS + n] = acc[jn][r] + mn;
    }
  }
}

// ---------------- launch ----------------
extern "C" void kernel_launch(void* const* d_in, const int* in_sizes, int n_in,
                              void* d_out, int out_size, void* d_ws, size_t ws_size,
                              hipStream_t stream) {
  const float* x      = (const float*)d_in[0];
  const float* rot    = (const float*)d_in[1];
  const float* mean   = (const float*)d_in[2];
  const float* whiten = (const float*)d_in[3];
  const float* cent   = (const float*)d_in[4];
  const int*   bits   = (const int*)d_in[5];

  char* ws = (char*)d_ws;
  unsigned short* g1h    = (unsigned short*)(ws + 0);        // 32 KB each
  unsigned short* g1l    = (unsigned short*)(ws + 32768);
  unsigned short* g2h    = (unsigned short*)(ws + 65536);
  unsigned short* g2l    = (unsigned short*)(ws + 98304);
  float*          bounds = (float*)(ws + 131072);            // 32 KB
  float*          invws  = (float*)(ws + 163840);            // 512 B

  const long nrows = (long)in_sizes[0] / DIMS;               // 524288
  float* out_x   = (float*)d_out;
  int*   out_idx = (int*)d_out + nrows * DIMS;

  pcaq_prep<<<64, NTHREADS, 0, stream>>>(rot, cent, bits, whiten,
                                         g1h, g1l, g2h, g2l, bounds, invws);
  pcaq_main<<<(unsigned)(nrows / ROWS_PER_BLOCK), NTHREADS, 0, stream>>>(
      x, mean, whiten, cent, g1h, g1l, g2h, g2l, bounds, invws, out_x, out_idx);
}
